// EnvelopeDetector_7189775253736
// MI455X (gfx1250) — matrix operand order repaired and checked by reading, not compiled, not run on hardware
//
#include <hip/hip_runtime.h>

typedef __attribute__((ext_vector_type(16))) _Float16 v16h;
typedef __attribute__((ext_vector_type(8)))  float    v8f;

#define BB  32
#define CC  64
#define TT  20000
#define K1  100
#define T1  19901      // T - 99
#define K2  50
#define T2  19852      // T1 - 49
#define T1P 19968      // padded y row length (multiple of 16, >= 19904 window reach)
#define SLAB_TILES 32
#define SLAB_OUT   (SLAB_TILES * 16)   // 512 outputs per workgroup slab
#define NTHREADS   256                 // 8 wave32

#define W1 (SLAB_OUT + 128)   // 640 staged x cols (K1 padded to 128)
#define S1 648                // LDS stride (halves): 1296 B = 324 words == 4 mod 64
#define W2 (SLAB_OUT + 96)    // 608 staged n cols (K2 padded to 96)
#define S2 616                // 1232 B = 308 words == 52 mod 64

__global__ void k_zero(float* stats) {
  int i = threadIdx.x;
  if (i < 2 * CC) stats[i] = 0.f;
}

__global__ __launch_bounds__(NTHREADS)
void k_conv1(const float* __restrict__ x, const float* __restrict__ wband,
             _Float16* __restrict__ y, float* __restrict__ stats)
{
  const int slab = blockIdx.x, g = blockIdx.y, c = blockIdx.z;
  const int ts  = slab * SLAB_OUT;
  const int tid = threadIdx.x;

  __shared__ _Float16 xs[16][S1];
  __shared__ float    wsh[K1];
  __shared__ float    red[2 * NTHREADS];

  if (tid < K1) wsh[tid] = wband[c * K1 + tid];

  // Stage x strip -> LDS as f16 (2 cols per thread-iteration, coalesced b64 loads)
  for (int idx = tid; idx < 16 * (W1 / 2); idx += NTHREADS) {
    int row = idx / (W1 / 2);
    int col = (idx % (W1 / 2)) * 2;
    long base = ((long)(g * 16 + row) * CC + c) * TT;
    int t0c = ts + col;
    float a = (t0c     < TT) ? x[base + t0c    ] : 0.f;
    float b = (t0c + 1 < TT) ? x[base + t0c + 1] : 0.f;
    xs[row][col]     = (_Float16)a;
    xs[row][col + 1] = (_Float16)b;
  }
  __syncthreads();

  const int lane  = tid & 31;
  const int wave  = tid >> 5;
  const int lhalf = lane >> 4;
  const int lrow  = lane & 15;

  // Banded-Toeplitz A fragments (contiguous K order: lane-half splits K 0-15 vs 16-31, matching B)
  v16h A[4];
#pragma unroll
  for (int ch = 0; ch < 4; ++ch) {
#pragma unroll
    for (int e = 0; e < 16; ++e) {
      int k = ch * 32 + 16 * lhalf + e;
      int d = k - lrow;                         // w index = K - M
      A[ch][e] = (_Float16)((d >= 0 && d < K1) ? wsh[d] : 0.f);
    }
  }

  float sum = 0.f, ssq = 0.f;

#pragma unroll
  for (int tt = 0; tt < SLAB_TILES / 8; ++tt) { // 4 tiles per wave
    const int tile = wave + 8 * tt;
    const int t0   = ts + tile * 16;
    v8f acc = {};
#pragma unroll
    for (int ch = 0; ch < 4; ++ch) {
      union { uint4 q[2]; v16h v; } u;          // B frag: 16 consecutive halves per lane
      const uint4* p = (const uint4*)&xs[lrow][tile * 16 + ch * 32 + 16 * lhalf];
      u.q[0] = p[0]; u.q[1] = p[1];
      acc = __builtin_amdgcn_wmma_f32_16x16x32_f16(false, A[ch], false, u.v,
                                                   (short)0, acc, false, false);
    }
    const int tbase = t0 + 8 * lhalf;           // D row = VGPR + 8*lane_half
#pragma unroll
    for (int v = 0; v < 8; ++v) {               // masked BN partial sums
      float dv = (tbase + v < T1) ? acc[v] : 0.f;
      sum += dv; ssq += dv * dv;
    }
    long yrow = ((long)(g * 16 + lrow) * CC + c) * T1P;
    if (tbase + 7 < T1) {
      union { _Float16 h[8]; uint4 q; } s;
#pragma unroll
      for (int v = 0; v < 8; ++v) s.h[v] = (_Float16)acc[v];
      *(uint4*)&y[yrow + tbase] = s.q;          // 16B store, aligned
    } else {
#pragma unroll
      for (int v = 0; v < 8; ++v) {             // zero-fill pad region
        int t = tbase + v;
        if (t < T1P) y[yrow + t] = (_Float16)((t < T1) ? acc[v] : 0.f);
      }
    }
  }

  // Block reduction -> 2 atomics per workgroup
  red[tid] = sum; red[NTHREADS + tid] = ssq;
  __syncthreads();
  for (int s = NTHREADS / 2; s > 0; s >>= 1) {
    if (tid < s) {
      red[tid] += red[tid + s];
      red[NTHREADS + tid] += red[NTHREADS + tid + s];
    }
    __syncthreads();
  }
  if (tid == 0) {
    atomicAdd(&stats[c],      red[0]);
    atomicAdd(&stats[CC + c], red[NTHREADS]);
  }
}

__global__ void k_finalize(const float* __restrict__ stats,
                           const float* __restrict__ gamma,
                           const float* __restrict__ beta,
                           float* __restrict__ scsh)
{
  int c = threadIdx.x;
  if (c >= CC) return;
  const float n = (float)BB * (float)T1;
  float mean = stats[c] / n;
  float var  = stats[CC + c] / n - mean * mean;
  float r    = rsqrtf(var + 1e-5f);
  float scale = gamma[c] * r;
  scsh[2 * c]     = scale;
  scsh[2 * c + 1] = beta[c] - mean * scale;
}

__global__ __launch_bounds__(NTHREADS)
void k_conv2(const _Float16* __restrict__ y, const float* __restrict__ wlow,
             const float* __restrict__ blow, const float* __restrict__ scsh,
             float* __restrict__ z)
{
  const int slab = blockIdx.x, g = blockIdx.y, c = blockIdx.z;
  const int ts  = slab * SLAB_OUT;
  const int tid = threadIdx.x;

  __shared__ _Float16 ns[16][S2];
  __shared__ float    wsh[K2];

  if (tid < K2) wsh[tid] = wlow[c * K2 + tid];
  const float scale = scsh[2 * c], shift = scsh[2 * c + 1];

  // Stage y -> normalize -> abs -> f16 LDS
  for (int idx = tid; idx < 16 * (W2 / 2); idx += NTHREADS) {
    int row = idx / (W2 / 2);
    int col = (idx % (W2 / 2)) * 2;
    long base = ((long)(g * 16 + row) * CC + c) * T1P;
    int t0c = ts + col;
    float a = (t0c     < T1P) ? (float)y[base + t0c    ] : 0.f;
    float b = (t0c + 1 < T1P) ? (float)y[base + t0c + 1] : 0.f;
    ns[row][col]     = (_Float16)fabsf(a * scale + shift);
    ns[row][col + 1] = (_Float16)fabsf(b * scale + shift);
  }
  __syncthreads();

  const int lane  = tid & 31;
  const int wave  = tid >> 5;
  const int lhalf = lane >> 4;
  const int lrow  = lane & 15;

  v16h A[3];
#pragma unroll
  for (int ch = 0; ch < 3; ++ch) {
#pragma unroll
    for (int e = 0; e < 16; ++e) {
      int k = ch * 32 + 16 * lhalf + e;
      int d = k - lrow;
      A[ch][e] = (_Float16)((d >= 0 && d < K2) ? wsh[d] : 0.f);
    }
  }

  const float bias = blow[c];

#pragma unroll
  for (int tt = 0; tt < SLAB_TILES / 8; ++tt) {
    const int tile = wave + 8 * tt;
    const int t0   = ts + tile * 16;
    v8f acc;
#pragma unroll
    for (int v = 0; v < 8; ++v) acc[v] = bias;  // fold bias into C operand
#pragma unroll
    for (int ch = 0; ch < 3; ++ch) {
      union { uint4 q[2]; v16h v; } u;
      const uint4* p = (const uint4*)&ns[lrow][tile * 16 + ch * 32 + 16 * lhalf];
      u.q[0] = p[0]; u.q[1] = p[1];
      acc = __builtin_amdgcn_wmma_f32_16x16x32_f16(false, A[ch], false, u.v,
                                                   (short)0, acc, false, false);
    }
    const int tbase = t0 + 8 * lhalf;
    long zrow = ((long)(g * 16 + lrow) * CC + c) * T2;
    if (tbase + 7 < T2) {
      *(float4*)&z[zrow + tbase]     = make_float4(acc[0], acc[1], acc[2], acc[3]);
      *(float4*)&z[zrow + tbase + 4] = make_float4(acc[4], acc[5], acc[6], acc[7]);
    } else {
#pragma unroll
      for (int v = 0; v < 8; ++v) {
        int t = tbase + v;
        if (t < T2) z[zrow + t] = acc[v];
      }
    }
  }
}

extern "C" void kernel_launch(void* const* d_in, const int* in_sizes, int n_in,
                              void* d_out, int out_size, void* d_ws, size_t ws_size,
                              hipStream_t stream)
{
  const float* x     = (const float*)d_in[0];
  const float* wband = (const float*)d_in[1];
  const float* gamma = (const float*)d_in[2];
  const float* beta  = (const float*)d_in[3];
  const float* wlow  = (const float*)d_in[4];
  const float* blow  = (const float*)d_in[5];
  float* z = (float*)d_out;

  float*    stats = (float*)d_ws;            // [0..63]=sum, [64..127]=sumsq
  float*    scsh  = stats + 2 * CC;          // [128..255]=(scale,shift) pairs
  _Float16* y     = (_Float16*)((char*)d_ws + 1024);  // 32*64*T1P f16 ~ 82 MB

  const int slabs1 = (T1 + SLAB_OUT - 1) / SLAB_OUT;  // 39
  const int slabs2 = (T2 + SLAB_OUT - 1) / SLAB_OUT;  // 39

  k_zero<<<1, 128, 0, stream>>>(stats);
  k_conv1<<<dim3(slabs1, 2, CC), NTHREADS, 0, stream>>>(x, wband, y, stats);
  k_finalize<<<1, 64, 0, stream>>>(stats, gamma, beta, scsh);
  k_conv2<<<dim3(slabs2, 2, CC), NTHREADS, 0, stream>>>(y, wlow, blow, scsh, z);
}
